// Greedy_Base_hGLM_66941360276232
// MI455X (gfx1250) — matrix operand order repaired and checked by reading, not compiled, not run on hardware
//
#include <hip/hip_runtime.h>
#include <hip/hip_bf16.h>
#include <math.h>

#define SUB_NO 20
#define E_NO   2000
#define I_NO   500
#define T_NO_  200
#define T_DATA 20000

typedef __attribute__((ext_vector_type(16))) _Float16 v16h;
typedef __attribute__((ext_vector_type(8)))  float    v8f;
typedef __attribute__((ext_vector_type(4)))  float    v4f;

// ---------------------------------------------------------------------------
// Phase A1: Gumbel softmax over subunit axis -> f16 B matrix (32 rows x Kpad)
// Rows 20..31 and columns K..Kpad-1 are zero-filled (WMMA padding).
// ---------------------------------------------------------------------------
__global__ void __launch_bounds__(256) gumbel_softmax_kernel(
    const float* __restrict__ logit, const float* __restrict__ u,
    _Float16* __restrict__ B, int K, int Kpad)
{
  int col = blockIdx.x * blockDim.x + threadIdx.x;
  if (col >= Kpad) return;
  if (col >= K) {
    for (int s = 0; s < 32; ++s) B[s * Kpad + col] = (_Float16)0.0f;
    return;
  }
  float v[SUB_NO];
  float vmax = -3.0e38f;
  for (int s = 0; s < SUB_NO; ++s) {
    float g = -__logf(-__logf(u[s * K + col] + 1e-8f) + 1e-8f);
    float x = (logit[s * K + col] + g) * 10.0f;   // divide by GUMBEL_TAU = 0.1
    v[s] = x;
    vmax = fmaxf(vmax, x);
  }
  float sum = 0.0f;
  for (int s = 0; s < SUB_NO; ++s) { v[s] = __expf(v[s] - vmax); sum += v[s]; }
  float inv = 1.0f / sum;
  for (int s = 0; s < SUB_NO; ++s) B[s * Kpad + col] = (_Float16)(v[s] * inv);
  for (int s = SUB_NO; s < 32; ++s) B[s * Kpad + col] = (_Float16)0.0f;
}

// ---------------------------------------------------------------------------
// Phase A2: alpha synaptic kernels  kern[ei][s][d] = tt*exp(-tt)*W_syn[s][ei]
// ---------------------------------------------------------------------------
__global__ void __launch_bounds__(256) alpha_kernel(
    const float* __restrict__ W_syn, const float* __restrict__ Tau_syn,
    const float* __restrict__ Delta_syn, float* __restrict__ kern)
{
  int idx = blockIdx.x * blockDim.x + threadIdx.x;
  if (idx >= SUB_NO * 2 * T_NO_) return;
  int d  = idx % T_NO_;
  int ei = (idx / T_NO_) % 2;
  int s  = idx / (2 * T_NO_);
  float delta = __expf(Delta_syn[s * 2 + ei]);
  float tau   = __expf(Tau_syn[s * 2 + ei]);
  float tt    = fmaxf((float)d - delta, 0.0f) / tau;
  kern[ei * (SUB_NO * T_NO_) + s * T_NO_ + d] = tt * __expf(-tt) * W_syn[s * 2 + ei];
}

// ---------------------------------------------------------------------------
// Phase B: streaming WMMA GEMM   out(M x 20) = X(M x K, f32) * B^T
// B staged as f16 (32 x Kpad), zero padded. One wave per 32-row M block
// (two 16-row WMMA tiles): B fragments loaded once, reused by both A tiles
// -> 4x v_wmma_f32_16x16x32_f16 per 32-wide K step, half the B traffic.
// ---------------------------------------------------------------------------
__global__ void __launch_bounds__(256) wmma_gemm_kernel(
    const float* __restrict__ X, const _Float16* __restrict__ B,
    float* __restrict__ out, int M, int K, int Kpad)
{
  int pair = (int)((blockIdx.x * blockDim.x + threadIdx.x) >> 5);
  int lane = threadIdx.x & 31;
  if (pair * 32 >= M) return;                  // uniform per wave: EXEC all-ones
  int khalf = lane >> 4;                       // which K half this lane supplies
  int n0    = lane & 15;
  int mrow0 = pair * 32 + n0;                  // A tile 0 row
  bool has1 = (pair * 32 + 16) < M;            // uniform across wave
  int mrow1 = has1 ? (mrow0 + 16) : mrow0;     // A tile 1 row (safe fallback)
  const float*    xrow0 = X + (size_t)mrow0 * K;
  const float*    xrow1 = X + (size_t)mrow1 * K;
  const _Float16* brow0 = B + (size_t)n0 * Kpad;          // N = 0..15
  const _Float16* brow1 = B + (size_t)(n0 + 16) * Kpad;   // N = 16..31

  v8f acc00 = {}, acc01 = {}, acc10 = {}, acc11 = {};
  for (int kb = 0; kb < Kpad; kb += 32) {
    int k0 = kb + khalf * 16;
    int k1 = kb + khalf * 16 + 8;
    v16h a0, a1;
    if (kb + 32 <= K) {                        // fast path: vector loads + cvt
      v4f p0 = *reinterpret_cast<const v4f*>(xrow0 + k0);
      v4f p1 = *reinterpret_cast<const v4f*>(xrow0 + k0 + 4);
      v4f p2 = *reinterpret_cast<const v4f*>(xrow0 + k1);
      v4f p3 = *reinterpret_cast<const v4f*>(xrow0 + k1 + 4);
      v4f q0 = *reinterpret_cast<const v4f*>(xrow1 + k0);
      v4f q1 = *reinterpret_cast<const v4f*>(xrow1 + k0 + 4);
      v4f q2 = *reinterpret_cast<const v4f*>(xrow1 + k1);
      v4f q3 = *reinterpret_cast<const v4f*>(xrow1 + k1 + 4);
#pragma unroll
      for (int j = 0; j < 4; ++j) {
        a0[j]      = (_Float16)p0[j];
        a0[4 + j]  = (_Float16)p1[j];
        a0[8 + j]  = (_Float16)p2[j];
        a0[12 + j] = (_Float16)p3[j];
        a1[j]      = (_Float16)q0[j];
        a1[4 + j]  = (_Float16)q1[j];
        a1[8 + j]  = (_Float16)q2[j];
        a1[12 + j] = (_Float16)q3[j];
      }
      if (kb + 160 < K) {                      // WGP-scope stream prefetch
        __builtin_prefetch(xrow0 + kb + 160, 0, 3);
        __builtin_prefetch(xrow1 + kb + 160, 0, 3);
      }
    } else {                                   // K remainder: zero-guarded
#pragma unroll
      for (int j = 0; j < 8; ++j) {
        float xa = (k0 + j < K) ? xrow0[k0 + j] : 0.0f;
        float xb = (k1 + j < K) ? xrow0[k1 + j] : 0.0f;
        float ya = (k0 + j < K) ? xrow1[k0 + j] : 0.0f;
        float yb = (k1 + j < K) ? xrow1[k1 + j] : 0.0f;
        a0[j]     = (_Float16)xa;
        a0[8 + j] = (_Float16)xb;
        a1[j]     = (_Float16)ya;
        a1[8 + j] = (_Float16)yb;
      }
    }
    // B fragment: lane holds column N, contiguous K = khalf*16 .. +15 (32B load)
    v16h bb0 = *reinterpret_cast<const v16h*>(brow0 + kb + khalf * 16);
    v16h bb1 = *reinterpret_cast<const v16h*>(brow1 + kb + khalf * 16);
    acc00 = __builtin_amdgcn_wmma_f32_16x16x32_f16(false, a0, false, bb0,
                                                   (short)0, acc00, false, false);
    acc10 = __builtin_amdgcn_wmma_f32_16x16x32_f16(false, a1, false, bb0,
                                                   (short)0, acc10, false, false);
    acc01 = __builtin_amdgcn_wmma_f32_16x16x32_f16(false, a0, false, bb1,
                                                   (short)0, acc01, false, false);
    acc11 = __builtin_amdgcn_wmma_f32_16x16x32_f16(false, a1, false, bb1,
                                                   (short)0, acc11, false, false);
  }
  // D layout: lane -> N = lane&15, VGPR r -> M = 8*khalf + r
  int mbase0 = pair * 32 + 8 * khalf;
#pragma unroll
  for (int r = 0; r < 8; ++r)
    out[(size_t)(mbase0 + r) * SUB_NO + n0] = acc00[r];
  if (n0 < 4) {                                // only subunits 16..19 are real
#pragma unroll
    for (int r = 0; r < 8; ++r)
      out[(size_t)(mbase0 + r) * SUB_NO + 16 + n0] = acc01[r];
  }
  if (has1) {
    int mbase1 = mbase0 + 16;
#pragma unroll
    for (int r = 0; r < 8; ++r)
      out[(size_t)(mbase1 + r) * SUB_NO + n0] = acc10[r];
    if (n0 < 4) {
#pragma unroll
      for (int r = 0; r < 8; ++r)
        out[(size_t)(mbase1 + r) * SUB_NO + 16 + n0] = acc11[r];
    }
  }
}

// ---------------------------------------------------------------------------
// Phase C: depthwise causal conv, length-200 kernels.
// syn[t,s] = sum_d ke[s,d]*in_e[t-d,s] + ki[s,d]*in_i[t-d,s]
// Input window staged in LDS (327 x 20 x 2 floats = 52 KB).
// ---------------------------------------------------------------------------
__global__ void __launch_bounds__(256) conv_kernel(
    const float* __restrict__ in_e, const float* __restrict__ in_i,
    const float* __restrict__ kern, float* __restrict__ syn, int T)
{
  const int TT = 128;
  const int W  = TT + T_NO_ - 1;   // 327
  __shared__ float se[W * SUB_NO];
  __shared__ float si[W * SUB_NO];
  int t0 = blockIdx.x * TT;
  for (int idx = threadIdx.x; idx < W * SUB_NO; idx += blockDim.x) {
    int r = idx / SUB_NO, c = idx - r * SUB_NO;
    int t = t0 - (T_NO_ - 1) + r;
    float ve = 0.0f, vi = 0.0f;
    if (t >= 0 && t < T) { ve = in_e[t * SUB_NO + c]; vi = in_i[t * SUB_NO + c]; }
    se[idx] = ve; si[idx] = vi;
  }
  __syncthreads();
  const float* ke = kern;
  const float* ki = kern + SUB_NO * T_NO_;
  for (int o = threadIdx.x; o < TT * SUB_NO; o += blockDim.x) {
    int tt = o / SUB_NO, s = o - tt * SUB_NO;
    int t  = t0 + tt;
    if (t < T) {
      const float* kes = ke + s * T_NO_;
      const float* kis = ki + s * T_NO_;
      float acc = 0.0f;
#pragma unroll 4
      for (int d = 0; d < T_NO_; ++d) {
        int row = tt + (T_NO_ - 1) - d;        // x[t-d]
        acc = fmaf(se[row * SUB_NO + s], kes[d], acc);
        acc = fmaf(si[row * SUB_NO + s], kis[d], acc);
      }
      syn[t * SUB_NO + s] = acc;
    }
  }
}

// ---------------------------------------------------------------------------
// Phase D: per-timestep 20-node tree recursion (idx 19 -> 0), then output.
// ---------------------------------------------------------------------------
__global__ void __launch_bounds__(256) tree_kernel(
    const float* __restrict__ syn, const float* __restrict__ C_den,
    const float* __restrict__ W_sub, const float* __restrict__ V_o,
    float* __restrict__ out, int T)
{
  __shared__ float cw[SUB_NO * SUB_NO];
  __shared__ float w0v0[2];
  for (int idx = threadIdx.x; idx < SUB_NO * SUB_NO; idx += blockDim.x)
    cw[idx] = C_den[idx] * W_sub[idx % SUB_NO];
  if (threadIdx.x == 0) { w0v0[0] = W_sub[0]; w0v0[1] = V_o[0]; }
  __syncthreads();
  int t = blockIdx.x * blockDim.x + threadIdx.x;
  if (t >= T) return;
  float synv[SUB_NO], sub[SUB_NO];
#pragma unroll
  for (int s = 0; s < SUB_NO; ++s) { synv[s] = syn[t * SUB_NO + s]; sub[s] = 0.0f; }
#pragma unroll
  for (int idx = SUB_NO - 1; idx >= 0; --idx) {
    float acc = synv[idx];
#pragma unroll
    for (int j = 0; j < SUB_NO; ++j) acc = fmaf(cw[idx * SUB_NO + j], sub[j], acc);
    sub[idx] = tanhf(acc);
  }
  out[t] = sub[0] * w0v0[0] + w0v0[1];
}

// ---------------------------------------------------------------------------
extern "C" void kernel_launch(void* const* d_in, const int* in_sizes, int n_in,
                              void* d_out, int out_size, void* d_ws, size_t ws_size,
                              hipStream_t stream) {
  (void)in_sizes; (void)n_in; (void)out_size; (void)ws_size;
  const float* S_e    = (const float*)d_in[0];
  const float* S_i    = (const float*)d_in[1];
  const float* C_den  = (const float*)d_in[2];
  const float* u_e    = (const float*)d_in[3];
  const float* u_i    = (const float*)d_in[4];
  const float* W_syn  = (const float*)d_in[5];
  const float* Tau    = (const float*)d_in[6];
  const float* Delta  = (const float*)d_in[7];
  const float* W_sub  = (const float*)d_in[8];
  const float* V_o    = (const float*)d_in[9];
  const float* loge   = (const float*)d_in[10];
  const float* logi   = (const float*)d_in[11];
  float* out = (float*)d_out;

  // Workspace layout (256-aligned offsets), ~5.0 MB total
  char* ws = (char*)d_ws;
  const int KPE = 2016;                        // E_NO rounded up to 32
  const int KPI = 512;                         // I_NO rounded up to 32
  _Float16* BE   = (_Float16*)(ws + 0);        // 32*2016*2 = 129024 B
  _Float16* BI   = (_Float16*)(ws + 131072);   // 32*512*2  =  32768 B
  float*    kern = (float*)(ws + 196608);      // 2*20*200*4 = 32000 B
  float*    ine  = (float*)(ws + 262144);      // 20000*20*4 = 1.6 MB
  float*    ini  = (float*)(ws + 262144 + 1600000);
  float*    syn  = (float*)(ws + 262144 + 3200000);

  gumbel_softmax_kernel<<<(KPE + 255) / 256, 256, 0, stream>>>(loge, u_e, BE, E_NO, KPE);
  gumbel_softmax_kernel<<<(KPI + 255) / 256, 256, 0, stream>>>(logi, u_i, BI, I_NO, KPI);
  alpha_kernel<<<(SUB_NO * 2 * T_NO_ + 255) / 256, 256, 0, stream>>>(W_syn, Tau, Delta, kern);

  int pairs = (T_DATA + 31) / 32;              // 625 waves, 8 waves/block
  int gblk  = (pairs + 7) / 8;
  wmma_gemm_kernel<<<gblk, 256, 0, stream>>>(S_e, BE, ine, T_DATA, E_NO, KPE);
  wmma_gemm_kernel<<<gblk, 256, 0, stream>>>(S_i, BI, ini, T_DATA, I_NO, KPI);

  conv_kernel<<<(T_DATA + 127) / 128, 256, 0, stream>>>(ine, ini, kern, syn, T_DATA);
  tree_kernel<<<(T_DATA + 255) / 256, 256, 0, stream>>>(syn, C_den, W_sub, V_o, out, T_DATA);
}
